// OutlierPagedModel_53858889892013
// MI455X (gfx1250) — matrix operand order repaired and checked by reading, not compiled, not run on hardware
//
#include <hip/hip_runtime.h>
#include <hip/hip_bf16.h>
#include <math.h>

typedef _Float16 f16;
typedef __attribute__((ext_vector_type(16))) _Float16 v16h;
typedef __attribute__((ext_vector_type(8)))  float    v8f;

#define T_ 256
#define D_ 2048
#define I_ 5632
#define E_ 8
constexpr size_t P_ = ((size_t)I_ * D_ + 4) / 5;   // TQ1_0 packed count per matrix

// ---- workspace layout (bytes) ----
constexpr size_t XH_B    = (size_t)T_ * D_ * sizeof(f16);
constexpr size_t GATES_B = (size_t)T_ * E_ * sizeof(float);
constexpr size_t H_B     = (size_t)9 * T_ * I_ * sizeof(f16);
// y: 9 * T * D * f32 follows

// ---- async global->LDS helpers (CDNA5: tracked on ASYNCcnt) ----
static __device__ __forceinline__ void async_copy_b128(const void* gsrc, void* lds_dst) {
  unsigned lds = (unsigned)(uintptr_t)lds_dst;        // low bits = LDS byte address
  asm volatile("global_load_async_to_lds_b128 %0, %1, off"
               :: "v"(lds), "v"(gsrc) : "memory");
}
static __device__ __forceinline__ void wait_async0() {
#if __has_builtin(__builtin_amdgcn_s_wait_asynccnt)
  __builtin_amdgcn_s_wait_asynccnt(0);
#else
  asm volatile("s_wait_asynccnt 0x0" ::: "memory");
#endif
}

static __device__ __forceinline__ void build_lut(f16* sLut, int tid, int nthr) {
  // sLut[b*5+j] = (b / 3^j) % 3 - 1  for b in [0,243)
  for (int b = tid; b < 243; b += nthr) {
    sLut[b * 5 + 0] = (f16)(float)(b % 3 - 1);
    sLut[b * 5 + 1] = (f16)(float)((b / 3) % 3 - 1);
    sLut[b * 5 + 2] = (f16)(float)((b / 9) % 3 - 1);
    sLut[b * 5 + 3] = (f16)(float)((b / 27) % 3 - 1);
    sLut[b * 5 + 4] = (f16)(float)((b / 81) % 3 - 1);
  }
}

// dequant 16 consecutive elements of a ternary row into a v16h (32-bit index math only)
static __device__ __forceinline__ v16h deq16(const int* __restrict__ packed,
                                             const f16* __restrict__ sLut,
                                             unsigned nf) {
  v16h r;
#pragma unroll
  for (int s = 0; s < 16; ++s) {
    unsigned q  = nf / 5u;              // 32-bit magic mul
    unsigned m5 = nf - q * 5u;
    r[s] = sLut[(unsigned)packed[q] * 5u + m5];
    ++nf;
  }
  return r;
}

// ---------------------------------------------------------------- router
__global__ __launch_bounds__(256)
void k_router(const float* __restrict__ x, const float* __restrict__ rw,
              float* __restrict__ gates) {
  int t = threadIdx.x;
  if (t >= T_) return;
  float logits[E_];
  for (int e = 0; e < E_; ++e) {
    float s = 0.f;
    const float* xr = x + (size_t)t * D_;
    const float* wr = rw + (size_t)e * D_;
    for (int d = 0; d < D_; ++d) s += xr[d] * wr[d];
    logits[e] = s;
  }
  int i1 = 0; float v1 = logits[0];
  for (int e = 1; e < E_; ++e) if (logits[e] > v1) { v1 = logits[e]; i1 = e; }
  int i2 = -1; float v2 = -3.0e38f;
  for (int e = 0; e < E_; ++e) if (e != i1 && logits[e] > v2) { v2 = logits[e]; i2 = e; }
  float e1 = 1.f, e2 = __expf(v2 - v1);
  float inv = 1.f / (e1 + e2);
  for (int e = 0; e < E_; ++e)
    gates[t * E_ + e] = (e == i1) ? e1 * inv : (e == i2) ? e2 * inv : 0.f;
}

// ---------------------------------------------------------------- f32 -> f16 (vectorized)
__global__ __launch_bounds__(256)
void k_cvt(const float* __restrict__ x, f16* __restrict__ xh) {
  size_t i = ((size_t)blockIdx.x * 256 + threadIdx.x) * 4;
  float4 v = *(const float4*)(x + i);
  f16 o[4] = {(f16)v.x, (f16)v.y, (f16)v.z, (f16)v.w};
  *(float2*)(xh + i) = *(float2*)o;
}

// ---------------------------------------------------------------- gate/up GEMM + SwiGLU
// grid: (I/64, T/64, 9)  block: 128 (4 waves). wave w owns N-subtile w, loops 4 M-subtiles.
// slot z==8 is the shared INT8 expert.
__global__ __launch_bounds__(128)
void k_gateup(const f16* __restrict__ xh,
              const int* __restrict__ gate_packed, const int* __restrict__ up_packed,
              const float* __restrict__ gate_s, const float* __restrict__ up_s,
              const int* __restrict__ sh_gate_w, const int* __restrict__ sh_up_w,
              const float* __restrict__ sh_gate_s, const float* __restrict__ sh_up_s,
              f16* __restrict__ h) {
  const int n0 = blockIdx.x * 64;
  const int m0 = blockIdx.y * 64;
  const int e  = blockIdx.z;
  const int tid  = threadIdx.x;
  const int wave = tid >> 5;
  const int lane = tid & 31;

  // fragment-major tiles: [subtile][lane][16 f16] (32B contiguous per lane)
  __shared__ __align__(32) f16 sA[4 * 32 * 16];
  __shared__ __align__(32) f16 sG[4 * 32 * 16];
  __shared__ __align__(32) f16 sU[4 * 32 * 16];
  __shared__ f16 sLut[243 * 5];

  build_lut(sLut, tid, 128);
  __syncthreads();

  const size_t eP = (size_t)e * P_;
  v8f accG[4] = {}, accU[4] = {};

  // B staging task: row n = tid>>1 (0..63), K-half kc = (tid&1)*16
  const int bn  = tid >> 1;
  const int bkc = (tid & 1) * 16;
  const int bfrag = (bn >> 4) * 512 + (((bn & 15) + 16 * (bkc >> 4)) * 16);
  const unsigned bRowG = (unsigned)(n0 + bn) * D_;

  for (int k0 = 0; k0 < D_; k0 += 32) {
    // ---- stage A (64x32) via async global->LDS b128, fragment layout
#pragma unroll
    for (int c = 0; c < 2; ++c) {
      int ch  = tid + c * 128;           // 0..255
      int m   = ch >> 2, oct = ch & 3;   // row, K-octet
      int dst = (m >> 4) * 512 + (((m & 15) + 16 * (oct >> 1)) * 16) + 8 * (oct & 1);
      async_copy_b128(xh + (size_t)(m0 + m) * D_ + k0 + oct * 8, &sA[dst]);
    }
    // ---- stage B (gate & up, 64x32 each): 16 contiguous K per thread -> one 32B store
    if (e == 8) {
      const int* gw = sh_gate_w + (size_t)bRowG + k0 + bkc;
      const int* uw = sh_up_w   + (size_t)bRowG + k0 + bkc;
      v16h tg, tu;
#pragma unroll
      for (int s = 0; s < 16; ++s) { tg[s] = (f16)(float)gw[s]; tu[s] = (f16)(float)uw[s]; }
      *(v16h*)&sG[bfrag] = tg;
      *(v16h*)&sU[bfrag] = tu;
    } else {
      unsigned nf = bRowG + (unsigned)(k0 + bkc);
      *(v16h*)&sG[bfrag] = deq16(gate_packed + eP, sLut, nf);
      *(v16h*)&sU[bfrag] = deq16(up_packed + eP, sLut, nf);
    }
    wait_async0();
    __syncthreads();

    v16h bg = *(const v16h*)&sG[wave * 512 + lane * 16];
    v16h bu = *(const v16h*)&sU[wave * 512 + lane * 16];
#pragma unroll
    for (int ms = 0; ms < 4; ++ms) {
      v16h a = *(const v16h*)&sA[ms * 512 + lane * 16];
      accG[ms] = __builtin_amdgcn_wmma_f32_16x16x32_f16(false, a, false, bg, (short)0, accG[ms], false, false);
      accU[ms] = __builtin_amdgcn_wmma_f32_16x16x32_f16(false, a, false, bu, (short)0, accU[ms], false, false);
    }
    __syncthreads();
  }

  // ---- epilogue: scale, SiLU(g)*u, store f16 hidden
  const int gcol = n0 + wave * 16 + (lane & 15);
  float gs, us;
  if (e == 8) { gs = sh_gate_s[gcol]; us = sh_up_s[gcol]; }
  else        { gs = gate_s[(size_t)e * I_ + gcol]; us = up_s[(size_t)e * I_ + gcol]; }
#pragma unroll
  for (int ms = 0; ms < 4; ++ms) {
#pragma unroll
    for (int r = 0; r < 8; ++r) {
      int tok = m0 + ms * 16 + r + (lane >> 4) * 8;
      float g = accG[ms][r] * gs;
      float u = accU[ms][r] * us;
      float si = g / (1.f + __expf(-g));
      h[((size_t)e * T_ + tok) * I_ + gcol] = (f16)(si * u);
    }
  }
}

// ---------------------------------------------------------------- down GEMM
// grid: (D/64, T/64, 9)  block: 128
__global__ __launch_bounds__(128)
void k_down(const f16* __restrict__ h,
            const int* __restrict__ down_packed, const float* __restrict__ down_s,
            const int* __restrict__ sh_down_w, const float* __restrict__ sh_down_s,
            float* __restrict__ y) {
  const int d0 = blockIdx.x * 64;
  const int m0 = blockIdx.y * 64;
  const int e  = blockIdx.z;
  const int tid  = threadIdx.x;
  const int wave = tid >> 5;
  const int lane = tid & 31;

  __shared__ __align__(32) f16 sA[4 * 32 * 16];
  __shared__ __align__(32) f16 sW[4 * 32 * 16];
  __shared__ f16 sLut[243 * 5];

  build_lut(sLut, tid, 128);
  __syncthreads();

  const size_t eP = (size_t)e * P_;
  v8f acc[4] = {};

  const int bn  = tid >> 1;
  const int bkc = (tid & 1) * 16;
  const int bfrag = (bn >> 4) * 512 + (((bn & 15) + 16 * (bkc >> 4)) * 16);
  const unsigned bRowG = (unsigned)(d0 + bn) * I_;
  const f16* hbase = h + (size_t)e * T_ * I_;

  for (int k0 = 0; k0 < I_; k0 += 32) {
#pragma unroll
    for (int c = 0; c < 2; ++c) {
      int ch  = tid + c * 128;
      int m   = ch >> 2, oct = ch & 3;
      int dst = (m >> 4) * 512 + (((m & 15) + 16 * (oct >> 1)) * 16) + 8 * (oct & 1);
      async_copy_b128(hbase + (size_t)(m0 + m) * I_ + k0 + oct * 8, &sA[dst]);
    }
    if (e == 8) {
      const int* dw = sh_down_w + (size_t)bRowG + k0 + bkc;
      v16h tw;
#pragma unroll
      for (int s = 0; s < 16; ++s) tw[s] = (f16)(float)dw[s];
      *(v16h*)&sW[bfrag] = tw;
    } else {
      unsigned nf = bRowG + (unsigned)(k0 + bkc);
      *(v16h*)&sW[bfrag] = deq16(down_packed + eP, sLut, nf);
    }
    wait_async0();
    __syncthreads();

    v16h bw = *(const v16h*)&sW[wave * 512 + lane * 16];
#pragma unroll
    for (int ms = 0; ms < 4; ++ms) {
      v16h a = *(const v16h*)&sA[ms * 512 + lane * 16];
      acc[ms] = __builtin_amdgcn_wmma_f32_16x16x32_f16(false, a, false, bw, (short)0, acc[ms], false, false);
    }
    __syncthreads();
  }

  const int gd = d0 + wave * 16 + (lane & 15);
  float ds = (e == 8) ? sh_down_s[gd] : down_s[(size_t)e * D_ + gd];
#pragma unroll
  for (int ms = 0; ms < 4; ++ms) {
#pragma unroll
    for (int r = 0; r < 8; ++r) {
      int tok = m0 + ms * 16 + r + (lane >> 4) * 8;
      y[((size_t)e * T_ + tok) * D_ + gd] = acc[ms][r] * ds;
    }
  }
}

// ---------------------------------------------------------------- deterministic combine
__global__ __launch_bounds__(256)
void k_combine(const float* __restrict__ y, const float* __restrict__ gates,
               float* __restrict__ out) {
  size_t i = (size_t)blockIdx.x * 256 + threadIdx.x;
  int t = (int)(i / D_);
  int d = (int)(i % D_);
  float acc = y[((size_t)8 * T_ + t) * D_ + d];      // shared expert
#pragma unroll
  for (int e = 0; e < E_; ++e)
    acc += gates[t * E_ + e] * y[((size_t)e * T_ + t) * D_ + d];
  out[i] = acc;
}

// ---------------------------------------------------------------- launch
extern "C" void kernel_launch(void* const* d_in, const int* in_sizes, int n_in,
                              void* d_out, int out_size, void* d_ws, size_t ws_size,
                              hipStream_t stream) {
  const float* x          = (const float*)d_in[0];
  const float* router_w   = (const float*)d_in[1];
  const int*   gate_packed= (const int*)d_in[2];
  const int*   up_packed  = (const int*)d_in[3];
  const int*   down_packed= (const int*)d_in[4];
  const float* gate_s     = (const float*)d_in[5];
  const float* up_s       = (const float*)d_in[6];
  const float* down_s     = (const float*)d_in[7];
  const int*   sh_gate_w  = (const int*)d_in[8];
  const int*   sh_up_w    = (const int*)d_in[9];
  const int*   sh_down_w  = (const int*)d_in[10];
  const float* sh_gate_s  = (const float*)d_in[11];
  const float* sh_up_s    = (const float*)d_in[12];
  const float* sh_down_s  = (const float*)d_in[13];
  float* out = (float*)d_out;

  char* ws = (char*)d_ws;
  f16*   xh    = (f16*)ws;
  float* gates = (float*)(ws + XH_B);
  f16*   h     = (f16*)(ws + XH_B + GATES_B);
  float* y     = (float*)(ws + XH_B + GATES_B + H_B);

  k_cvt<<<(T_ * D_) / (256 * 4), 256, 0, stream>>>(x, xh);
  k_router<<<1, 256, 0, stream>>>(x, router_w, gates);

  dim3 g1(I_ / 64, T_ / 64, 9);
  k_gateup<<<g1, 128, 0, stream>>>(xh, gate_packed, up_packed, gate_s, up_s,
                                   sh_gate_w, sh_up_w, sh_gate_s, sh_up_s, h);

  dim3 g2(D_ / 64, T_ / 64, 9);
  k_down<<<g2, 128, 0, stream>>>(h, down_packed, down_s, sh_down_w, sh_down_s, y);

  k_combine<<<(T_ * D_) / 256, 256, 0, stream>>>(y, gates, out);
}
